// InformerTranverse_52922587021393
// MI455X (gfx1250) — matrix operand order repaired and checked by reading, not compiled, not run on hardware
//
#include <hip/hip_runtime.h>
#include <math.h>

#define D_MODEL 512
#define NHEAD   8
#define HDIM    64
#define EPSV    1e-5f

typedef __attribute__((ext_vector_type(16))) _Float16 v16h;
typedef __attribute__((ext_vector_type(8)))  _Float16 v8h;
typedef __attribute__((ext_vector_type(4)))  _Float16 v4h;
typedef __attribute__((ext_vector_type(8)))  float    v8f;

#define CDIV(a,b) (((a)+(b)-1)/(b))

// ---------------------------------------------------------------------------
// threefry2x32 (JAX-compatible, 20 rounds) — used on host (key split) + device
// ---------------------------------------------------------------------------
__host__ __device__ static inline unsigned rotl32(unsigned x, int n) {
  return (x << n) | (x >> (32 - n));
}
__host__ __device__ static inline void threefry2x32(unsigned k0, unsigned k1,
                                                    unsigned c0, unsigned c1,
                                                    unsigned& o0, unsigned& o1) {
  const unsigned ks2 = 0x1BD11BDAu ^ k0 ^ k1;
  unsigned x0 = c0 + k0, x1 = c1 + k1;
  const int ra[4] = {13, 15, 26, 6};
  const int rb[4] = {17, 29, 16, 24};
  for (int i = 0; i < 4; ++i) { x0 += x1; x1 = rotl32(x1, ra[i]); x1 ^= x0; }
  x0 += k1; x1 += ks2 + 1u;
  for (int i = 0; i < 4; ++i) { x0 += x1; x1 = rotl32(x1, rb[i]); x1 ^= x0; }
  x0 += ks2; x1 += k0 + 2u;
  for (int i = 0; i < 4; ++i) { x0 += x1; x1 = rotl32(x1, ra[i]); x1 ^= x0; }
  x0 += k0; x1 += k1 + 3u;
  for (int i = 0; i < 4; ++i) { x0 += x1; x1 = rotl32(x1, rb[i]); x1 ^= x0; }
  x0 += k1; x1 += ks2 + 4u;
  for (int i = 0; i < 4; ++i) { x0 += x1; x1 = rotl32(x1, ra[i]); x1 ^= x0; }
  x0 += ks2; x1 += k0 + 5u;
  o0 = x0; o1 = x1;
}

// ---------------------------------------------------------------------------
// WMMA GEMM:  C[M,N] = act(scale * A[M,K] @ B + bias) + resid + pe
//   OPB==0: B is [K,N]; OPB==1: B is [N,K]. Batched via blockIdx.z.
// CONTRACT (host call sites): K % 32 == 0, N % 64 == 0; only M may be ragged
// (clamp+mask on A rows, store guard on C rows).
// Software-pipelined: global float4 loads for panel kb+32 are issued before
// the ds_load/wmma phase of panel kb, so VMEM overlaps the matrix math.
// ---------------------------------------------------------------------------
#define TBM 64
#define TBN 64
#define TBK 32
#define LDP 40  // LDS row stride in halves (80B: 16B-aligned rows, bank skew)

template <int OPB>
__global__ __launch_bounds__(128) void k_gemm(
    const float* __restrict__ A, const float* __restrict__ B,
    const float* __restrict__ bias, const float* __restrict__ resid,
    float* __restrict__ C,
    int M, int N, int K, int lda, int ldb, int ldc,
    long long sA, long long sB, long long sC,
    int act, int addpe, float scale)
{
  __shared__ _Float16 As[TBM * LDP];
  __shared__ _Float16 Bs[TBN * LDP];

  const int z = blockIdx.z;
  A += (long long)z * sA;
  B += (long long)z * sB;
  C += (long long)z * sC;

  const int m0 = blockIdx.y * TBM;
  const int n0 = blockIdx.x * TBN;
  const int tid = threadIdx.x;
  const int lane = tid & 31;
  const int wave = tid >> 5;
  const int wr = (wave >> 1) * 32;
  const int wc = (wave & 1) * 32;
  const int lhalf = lane >> 4;
  const int l16 = lane & 15;

  // ---- per-thread loop-invariant fill descriptors ----
  const float* aPtr[4];  float aMsk[4];  int aLds[4];
  const float* bPtr[4];  int bLds[4];
  #pragma unroll
  for (int it = 0; it < 4; ++it) {
    int i = tid + it * 128;
    {   // A: 64 rows x 8 float4 per row
      int r = i >> 3, c4 = (i & 7) * 4;
      int gm = m0 + r;
      int cg = gm < M ? gm : (M - 1);
      aPtr[it] = A + (long long)cg * lda + c4;
      aMsk[it] = gm < M ? 1.f : 0.f;
      aLds[it] = r * LDP + c4;
    }
    if (OPB) {  // B[N,K]: K-contiguous rows
      int r = i >> 3, c4 = (i & 7) * 4;
      bPtr[it] = B + (long long)(n0 + r) * ldb + c4;
      bLds[it] = r * LDP + c4;
    } else {    // B[K,N]: N-contiguous rows, transposed into LDS
      int kk = i >> 4, n4 = (i & 15) * 4;
      bPtr[it] = B + (long long)kk * ldb + (n0 + n4);
      bLds[it] = n4 * LDP + kk;
    }
  }

  v8f acc[2][2] = {};
  float4 ra[4], rb[4];

  // ---- prologue: load panel 0 into registers ----
  #pragma unroll
  for (int it = 0; it < 4; ++it) ra[it] = *(const float4*)(aPtr[it]);
  #pragma unroll
  for (int it = 0; it < 4; ++it) rb[it] = *(const float4*)(bPtr[it]);

  for (int kb = 0; kb < K; kb += TBK) {
    // ---- registers -> LDS (single loadcnt wait covers all 8 loads) ----
    #pragma unroll
    for (int it = 0; it < 4; ++it) {
      float m = aMsk[it];
      v4h hv = { (_Float16)(ra[it].x * m), (_Float16)(ra[it].y * m),
                 (_Float16)(ra[it].z * m), (_Float16)(ra[it].w * m) };
      *(v4h*)(As + aLds[it]) = hv;
    }
    #pragma unroll
    for (int it = 0; it < 4; ++it) {
      if (OPB) {
        v4h hv = { (_Float16)rb[it].x, (_Float16)rb[it].y,
                   (_Float16)rb[it].z, (_Float16)rb[it].w };
        *(v4h*)(Bs + bLds[it]) = hv;
      } else {
        Bs[bLds[it]          ] = (_Float16)rb[it].x;
        Bs[bLds[it] + LDP    ] = (_Float16)rb[it].y;
        Bs[bLds[it] + 2 * LDP] = (_Float16)rb[it].z;
        Bs[bLds[it] + 3 * LDP] = (_Float16)rb[it].w;
      }
    }
    __syncthreads();

    // ---- issue next panel's global loads (overlap with wmma below) ----
    int kn = kb + TBK;
    if (kn < K) {
      #pragma unroll
      for (int it = 0; it < 4; ++it) ra[it] = *(const float4*)(aPtr[it] + kn);
      #pragma unroll
      for (int it = 0; it < 4; ++it)
        rb[it] = OPB ? *(const float4*)(bPtr[it] + kn)
                     : *(const float4*)(bPtr[it] + (long long)kn * ldb);
    }

    // ---- fragment loads (ISA VGPR layouts, 128b LDS loads) + WMMA ----
    v16h af[2], bf[2];
    #pragma unroll
    for (int i = 0; i < 2; ++i) {
      const _Float16* p = &As[(wr + 16 * i + l16) * LDP + lhalf * 8];
      v8h lo = *(const v8h*)p;            // K = c8 .. c8+7     (c8 = lhalf*8)
      v8h hi = *(const v8h*)(p + 16);     // K = 16+c8 .. 16+c8+7
      af[i] = __builtin_shufflevector(lo, hi, 0,1,2,3,4,5,6,7,8,9,10,11,12,13,14,15);
    }
    #pragma unroll
    for (int j = 0; j < 2; ++j) {
      const _Float16* p = &Bs[(wc + 16 * j + l16) * LDP + lhalf * 8];
      v8h lo = *(const v8h*)p;            // K = c8 .. c8+7     (c8 = lhalf*8)
      v8h hi = *(const v8h*)(p + 16);     // K = 16+c8 .. 16+c8+7
      bf[j] = __builtin_shufflevector(lo, hi, 0,1,2,3,4,5,6,7,8,9,10,11,12,13,14,15);
    }
    #pragma unroll
    for (int i = 0; i < 2; ++i)
      #pragma unroll
      for (int j = 0; j < 2; ++j)
        acc[i][j] = __builtin_amdgcn_wmma_f32_16x16x32_f16(
            false, af[i], false, bf[j], (short)0, acc[i][j], false, false);
    __syncthreads();
  }

  // ---- epilogue (N full by contract; only M store guard) ----
  #pragma unroll
  for (int i = 0; i < 2; ++i) {
    #pragma unroll
    for (int j = 0; j < 2; ++j) {
      #pragma unroll
      for (int e = 0; e < 8; ++e) {
        int row = m0 + wr + 16 * i + e + lhalf * 8;
        int col = n0 + wc + 16 * j + l16;
        if (row < M) {
          float v = acc[i][j][e] * scale;
          if (bias) v += bias[col];
          if (act == 1) v = 0.5f * v * (1.f + erff(v * 0.70710678118f)); // exact GELU
          if (resid) v += resid[(long long)row * ldc + col];
          if (addpe) {  // sinusoidal positional embedding, d_model = 512
            float fr = expf(-(float)(col & ~1) * 0.017988946039f); // ln(1e4)/512
            v += (col & 1) ? cosf((float)row * fr) : sinf((float)row * fr);
          }
          C[(long long)row * ldc + col] = v;
        }
      }
    }
  }
}

// ---------------------------------------------------------------------------
// Elementwise / reduction kernels
// ---------------------------------------------------------------------------
// (B,24,F) -> [L, F*24] with c = f*24 + t  (the transpose(0,2,1).reshape flatten)
__global__ void k_flatten(const float* __restrict__ in, float* __restrict__ out,
                          int L, int F) {
  int i = blockIdx.x * 256 + threadIdx.x;
  int total = L * F * 24;
  if (i >= total) return;
  int l = i / (F * 24), r = i % (F * 24);
  int f = r / 24, t = r % 24;
  out[i] = in[(l * 24 + t) * F + f];
}

// circular 3-tap augmentation: aug[l][c*3+j] = x[(l-1+j) mod L][c]
__global__ void k_aug3(const float* __restrict__ x, float* __restrict__ aug,
                       int L, int C) {
  int i = blockIdx.x * 256 + threadIdx.x;
  int total = L * C * 3;
  if (i >= total) return;
  int l = i / (C * 3), r = i % (C * 3);
  int c = r / 3, j = r % 3;
  int src = (l - 1 + j + L) % L;
  aug[i] = x[src * C + c];
}

// LayerNorm over 512 features (one block per row)
__global__ __launch_bounds__(256) void k_ln(const float* __restrict__ x,
                                            const float* __restrict__ g,
                                            const float* __restrict__ b,
                                            float* __restrict__ y) {
  __shared__ float red[256];
  int l = blockIdx.x, t = threadIdx.x;
  const float* row = x + (long long)l * D_MODEL;
  float a0 = row[t], a1 = row[t + 256];
  red[t] = a0 + a1;
  __syncthreads();
  for (int o = 128; o > 0; o >>= 1) { if (t < o) red[t] += red[t + o]; __syncthreads(); }
  float mean = red[0] * (1.f / D_MODEL);
  __syncthreads();
  float d0 = a0 - mean, d1 = a1 - mean;
  red[t] = d0 * d0 + d1 * d1;
  __syncthreads();
  for (int o = 128; o > 0; o >>= 1) { if (t < o) red[t] += red[t + o]; __syncthreads(); }
  float rstd = rsqrtf(red[0] * (1.f / D_MODEL) + EPSV);
  y[(long long)l * D_MODEL + t]       = d0 * rstd * g[t]       + b[t];
  y[(long long)l * D_MODEL + t + 256] = d1 * rstd * g[t + 256] + b[t + 256];
}

// row softmax, optional causal limit per row (cols <= thr[row] allowed)
__global__ __launch_bounds__(256) void k_softmax(float* __restrict__ s, int N,
                                                 const int* __restrict__ thr) {
  __shared__ float red[256];
  int r = blockIdx.x, t = threadIdx.x;
  float* row = s + (long long)r * N;
  int lim = thr ? (thr[r] + 1) : N;
  float mx = -1e30f;
  for (int i = t; i < N; i += 256) { float v = (i < lim) ? row[i] : -1e30f; mx = fmaxf(mx, v); }
  red[t] = mx; __syncthreads();
  for (int o = 128; o > 0; o >>= 1) { if (t < o) red[t] = fmaxf(red[t], red[t + o]); __syncthreads(); }
  mx = red[0]; __syncthreads();
  float sum = 0.f;
  for (int i = t; i < N; i += 256) {
    float v = (i < lim) ? expf(row[i] - mx) : 0.f;
    row[i] = v; sum += v;
  }
  red[t] = sum; __syncthreads();
  for (int o = 128; o > 0; o >>= 1) { if (t < o) red[t] += red[t + o]; __syncthreads(); }
  float inv = 1.f / red[0];
  for (int i = t; i < N; i += 256) row[i] *= inv;
}

// JAX randint(key,(Lq,U),0,span) — counter split-halves convention
__global__ void k_randidx(int* __restrict__ idx, int n, unsigned k0, unsigned k1,
                          int span) {
  int i = blockIdx.x * 256 + threadIdx.x;
  if (i >= n) return;
  int half = n >> 1;
  unsigned c0, c1;
  if (i < half) { c0 = (unsigned)i; c1 = (unsigned)(i + half); }
  else          { c0 = (unsigned)(i - half); c1 = (unsigned)i; }
  unsigned o0, o1;
  threefry2x32(k0, k1, c0, c1, o0, o1);
  unsigned bits = (i < half) ? o0 : o1;
  idx[i] = (int)(bits % (unsigned)span);
}

// sparsity measure: M[h,l] = max_u(q_l . k_idx[l,u]) - sum_u(...)/Lk
__global__ __launch_bounds__(64) void k_sample_m(const float* __restrict__ q,
                                                 const float* __restrict__ k,
                                                 const int* __restrict__ idx,
                                                 float* __restrict__ Mout,
                                                 int Lq, int Lk, int U) {
  __shared__ float qs[HDIM];
  __shared__ float rmx[64], rsm[64];
  int l = blockIdx.x, h = blockIdx.y, t = threadIdx.x;
  qs[t] = q[(long long)l * D_MODEL + h * HDIM + t];
  __syncthreads();
  float mx = -1e30f, sm = 0.f;
  for (int u = t; u < U; u += 64) {
    const float* kr = k + (long long)idx[l * U + u] * D_MODEL + h * HDIM;
    float d = 0.f;
    #pragma unroll 8
    for (int e = 0; e < HDIM; ++e) d += qs[e] * kr[e];
    mx = fmaxf(mx, d); sm += d;
  }
  rmx[t] = mx; rsm[t] = sm; __syncthreads();
  for (int o = 32; o > 0; o >>= 1) {
    if (t < o) { rmx[t] = fmaxf(rmx[t], rmx[t + o]); rsm[t] += rsm[t + o]; }
    __syncthreads();
  }
  if (t == 0) Mout[(long long)h * Lq + l] = rmx[0] - rsm[0] / (float)Lk;
}

// iterative top-u argmax per head (descending, lowest-index tiebreak)
__global__ __launch_bounds__(256) void k_topk(const float* __restrict__ Msrc,
                                              float* __restrict__ Mw,
                                              int* __restrict__ mtop, int Lq, int u) {
  __shared__ float bv[256];
  __shared__ int   bi[256];
  int h = blockIdx.x, t = threadIdx.x;
  const float* src = Msrc + (long long)h * Lq;
  float* w = Mw + (long long)h * Lq;
  for (int i = t; i < Lq; i += 256) w[i] = src[i];
  __syncthreads();
  for (int r = 0; r < u; ++r) {
    float best = -1e30f; int besti = 0x7fffffff;
    for (int i = t; i < Lq; i += 256) {
      float v = w[i];
      if (v > best || (v == best && i < besti)) { best = v; besti = i; }
    }
    bv[t] = best; bi[t] = besti; __syncthreads();
    for (int o = 128; o > 0; o >>= 1) {
      if (t < o && (bv[t + o] > bv[t] || (bv[t + o] == bv[t] && bi[t + o] < bi[t]))) {
        bv[t] = bv[t + o]; bi[t] = bi[t + o];
      }
      __syncthreads();
    }
    if (t == 0) { mtop[h * u + r] = bi[0]; w[bi[0]] = -1e30f; }
    __syncthreads();
  }
}

__global__ __launch_bounds__(64) void k_gather_q(const float* __restrict__ q,
                                                 const int* __restrict__ mtop,
                                                 float* __restrict__ qr, int u) {
  int r = blockIdx.x, h = blockIdx.y, t = threadIdx.x;
  qr[(h * u + r) * HDIM + t] =
      q[(long long)mtop[h * u + r] * D_MODEL + h * HDIM + t];
}

__global__ __launch_bounds__(64) void k_scatter(float* __restrict__ ctx,
                                                const int* __restrict__ mtop,
                                                const float* __restrict__ upd, int u) {
  int r = blockIdx.x, h = blockIdx.y, t = threadIdx.x;
  ctx[(long long)mtop[h * u + r] * D_MODEL + h * HDIM + t] =
      upd[(h * u + r) * HDIM + t];
}

__global__ __launch_bounds__(256) void k_colmean(const float* __restrict__ v,
                                                 float* __restrict__ means, int L) {
  __shared__ float red[256];
  int c = blockIdx.x, t = threadIdx.x;
  float s = 0.f;
  for (int l = t; l < L; l += 256) s += v[(long long)l * D_MODEL + c];
  red[t] = s; __syncthreads();
  for (int o = 128; o > 0; o >>= 1) { if (t < o) red[t] += red[t + o]; __syncthreads(); }
  if (t == 0) means[c] = red[0] / (float)L;
}

__global__ void k_bcast(const float* __restrict__ means, float* __restrict__ out,
                        int total) {
  int i = blockIdx.x * 256 + threadIdx.x;
  if (i < total) out[i] = means[i % D_MODEL];
}

// causal context: per-channel cumulative sum along sequence
__global__ void k_cumsum(const float* __restrict__ v, float* __restrict__ ctx, int L) {
  int c = blockIdx.x * blockDim.x + threadIdx.x;
  if (c >= D_MODEL) return;
  float s = 0.f;
  for (int l = 0; l < L; ++l) {
    s += v[(long long)l * D_MODEL + c];
    ctx[(long long)l * D_MODEL + c] = s;
  }
}

// Informer "mix": view (L,H,D) data as the flat memory of (H,L,D) reshaped (L,512)
__global__ void k_mix(const float* __restrict__ ctx, float* __restrict__ out, int L) {
  int i = blockIdx.x * 256 + threadIdx.x;
  if (i >= L * D_MODEL) return;
  int h = i / (L * HDIM);
  int rem = i % (L * HDIM);
  int lp = rem / HDIM, d = rem % HDIM;
  out[i] = ctx[(long long)lp * D_MODEL + h * HDIM + d];
}

__global__ __launch_bounds__(256) void k_colstats(const float* __restrict__ y,
                                                  float* __restrict__ stats, int L) {
  __shared__ float rs[256], rq[256];
  int c = blockIdx.x, t = threadIdx.x;
  float s = 0.f, q = 0.f;
  for (int l = t; l < L; l += 256) {
    float v = y[(long long)l * D_MODEL + c];
    s += v; q += v * v;
  }
  rs[t] = s; rq[t] = q; __syncthreads();
  for (int o = 128; o > 0; o >>= 1) {
    if (t < o) { rs[t] += rs[t + o]; rq[t] += rq[t + o]; }
    __syncthreads();
  }
  if (t == 0) {
    float m = rs[0] / (float)L;
    stats[c] = m;
    stats[D_MODEL + c] = rq[0] / (float)L - m * m;  // biased var
  }
}

__global__ void k_bn_elu(float* __restrict__ y, const float* __restrict__ stats,
                         const float* __restrict__ g, const float* __restrict__ beta,
                         int total) {
  int i = blockIdx.x * 256 + threadIdx.x;
  if (i >= total) return;
  int c = i % D_MODEL;
  float v = (y[i] - stats[c]) * rsqrtf(stats[D_MODEL + c] + EPSV) * g[c] + beta[c];
  y[i] = v > 0.f ? v : expm1f(v);
}

// maxpool window 3, stride 2, pad 1 along sequence
__global__ void k_maxpool(const float* __restrict__ y, float* __restrict__ out,
                          int L, int Lo) {
  int i = blockIdx.x * 256 + threadIdx.x;
  if (i >= Lo * D_MODEL) return;
  int lo = i / D_MODEL, c = i % D_MODEL;
  float m = -1e30f;
  for (int j = -1; j <= 1; ++j) {
    int l = 2 * lo + j;
    if (l >= 0 && l < L) m = fmaxf(m, y[(long long)l * D_MODEL + c]);
  }
  out[i] = m;
}

// final projection + skip head: out[l] = skip(concat(dec@pw+pb, xd[:,:24]))
__global__ __launch_bounds__(256) void k_proj(const float* __restrict__ dec,
                                              const float* __restrict__ pw,
                                              const float* __restrict__ pb,
                                              const float* __restrict__ xd,
                                              const float* __restrict__ sw,
                                              const float* __restrict__ sb,
                                              float* __restrict__ out) {
  __shared__ float red[256];
  int l = blockIdx.x, t = threadIdx.x;
  const float* row = dec + (long long)l * D_MODEL;
  red[t] = row[t] * pw[t] + row[t + 256] * pw[t + 256];
  __syncthreads();
  for (int o = 128; o > 0; o >>= 1) { if (t < o) red[t] += red[t + o]; __syncthreads(); }
  if (t == 0) {
    float o0 = red[0] + pb[0];
    float r = o0 * sw[0];
    for (int i = 0; i < 24; ++i) r += xd[l * 192 + i] * sw[1 + i];
    out[l] = r + sb[0];
  }
}

// ---------------------------------------------------------------------------
// Host helpers
// ---------------------------------------------------------------------------
static void gemm(hipStream_t st, const float* A, const float* B, const float* bias,
                 const float* resid, float* C, int M, int N, int K,
                 int lda, int ldb, int ldc,
                 long long sA, long long sB, long long sC,
                 int opB, int act, int addpe, float scale, int batch) {
  dim3 g((unsigned)CDIV(N, TBN), (unsigned)CDIV(M, TBM), (unsigned)batch);
  if (opB)
    k_gemm<1><<<g, dim3(128), 0, st>>>(A, B, bias, resid, C, M, N, K, lda, ldb, ldc,
                                       sA, sB, sC, act, addpe, scale);
  else
    k_gemm<0><<<g, dim3(128), 0, st>>>(A, B, bias, resid, C, M, N, K, lda, ldb, ldc,
                                       sA, sB, sC, act, addpe, scale);
}

static int probU(int L) {
  int c = (int)ceil(log((double)L));
  int u = 5 * c;
  return u < L ? u : L;
}

// ---------------------------------------------------------------------------
// kernel_launch — full Informer forward.
// Parameter order assumes jax-tree (sorted dict key) flattening of params.
// ---------------------------------------------------------------------------
extern "C" void kernel_launch(void* const* d_in, const int* in_sizes, int n_in,
                              void* d_out, int out_size, void* d_ws, size_t ws_size,
                              hipStream_t stream) {
  (void)in_sizes; (void)n_in; (void)out_size; (void)ws_size;
#define IN(i) ((const float*)d_in[i])

  // ---- workspace carve ----
  char* base = (char*)d_ws;
  size_t cur = 0;
  auto alloc = [&](size_t nbytes) -> void* {
    void* p = base + cur;
    cur = (cur + nbytes + 255) & ~(size_t)255;
    return p;
  };
  auto af = [&](size_t nfloats) -> float* { return (float*)alloc(nfloats * 4); };

  const int L0 = 4096;
  float* XE   = af((size_t)L0 * 192);
  float* XME  = af((size_t)L0 * 96);
  float* XD   = af((size_t)L0 * 192);
  float* XMD  = af((size_t)L0 * 96);
  float* ENCX = af((size_t)L0 * D_MODEL);
  float* ENCO = af((size_t)L0 * D_MODEL);
  float* DECX = af((size_t)L0 * D_MODEL);
  float* T1   = af((size_t)L0 * D_MODEL);
  float* T2   = af((size_t)L0 * D_MODEL);
  float* Qb   = af((size_t)L0 * D_MODEL);
  float* Kb   = af((size_t)L0 * D_MODEL);
  float* Vb   = af((size_t)L0 * D_MODEL);
  float* AUG  = af((size_t)L0 * 1536);
  float* SC   = af((size_t)L0 * 1024);            // >= 8*45*4096 too
  float* UPD  = af(8 * 45 * HDIM);
  float* QR   = af(8 * 45 * HDIM);
  float* MM   = af(8 * (size_t)L0);
  float* MW   = af(8 * (size_t)L0);
  float* MEANS= af(D_MODEL);
  float* STATS= af(2 * D_MODEL);
  int* IDX  = (int*)alloc((size_t)L0 * 45 * 4);
  int* MTOP = (int*)alloc(512 * 4);

  // ---- JAX PRNG: split(key(42), 5) via threefry ----
  unsigned cat[10];
  {
    unsigned a[5], b[5];
    for (unsigned i = 0; i < 5; ++i) threefry2x32(0u, 42u, i, i + 5u, a[i], b[i]);
    for (int i = 0; i < 5; ++i) { cat[i] = a[i]; cat[5 + i] = b[i]; }
  }
  unsigned key0[5], key1[5];
  for (int i = 0; i < 5; ++i) { key0[i] = cat[2 * i]; key1[i] = cat[2 * i + 1]; }

  // ---- flatten inputs: (B,24,F) -> [L, F*24] ----
  k_flatten<<<CDIV(L0 * 192, 256), 256, 0, stream>>>(IN(0), XE, L0, 8);
  k_flatten<<<CDIV(L0 * 96, 256), 256, 0, stream>>>(IN(1), XME, L0, 4);
  k_flatten<<<CDIV(L0 * 192, 256), 256, 0, stream>>>(IN(2), XD, L0, 8);
  k_flatten<<<CDIV(L0 * 96, 256), 256, 0, stream>>>(IN(3), XMD, L0, 4);

  // ---- encoder embedding: conv(576) + temp(96) + pe ----
  k_aug3<<<CDIV(L0 * 576, 256), 256, 0, stream>>>(XE, AUG, L0, 192);
  gemm(stream, XME, IN(71), IN(70), nullptr, T1, L0, 512, 96, 96, 512, 512,
       0, 0, 0, 0, 0, 0, 1.f, 1);
  gemm(stream, AUG, IN(69), nullptr, T1, ENCX, L0, 512, 576, 576, 512, 512,
       0, 0, 0, 1, 0, 1, 1.f, 1);

  // ---- encoder stack ----
  int L = L0;
  for (int li = 0; li < 3; ++li) {
    int pb = 72 + 16 * li;
    const float *bk = IN(pb + 0), *bo = IN(pb + 1), *bq = IN(pb + 2), *bv = IN(pb + 3);
    const float *wk = IN(pb + 4), *wo = IN(pb + 5), *wq = IN(pb + 6), *wv = IN(pb + 7);
    const float *b1 = IN(pb + 8), *b2 = IN(pb + 9);
    const float *ln1b = IN(pb + 10), *ln1g = IN(pb + 11);
    const float *ln2b = IN(pb + 12), *ln2g = IN(pb + 13);
    const float *w1 = IN(pb + 14), *w2 = IN(pb + 15);

    gemm(stream, ENCX, wq, bq, nullptr, Qb, L, 512, 512, 512, 512, 512, 0,0,0, 0,0,0, 1.f, 1);
    gemm(stream, ENCX, wk, bk, nullptr, Kb, L, 512, 512, 512, 512, 512, 0,0,0, 0,0,0, 1.f, 1);
    gemm(stream, ENCX, wv, bv, nullptr, Vb, L, 512, 512, 512, 512, 512, 0,0,0, 0,0,0, 1.f, 1);

    int U = probU(L);  // 45 / 40 / 35
    k_randidx<<<CDIV(L * U, 256), 256, 0, stream>>>(IDX, L * U, key0[li], key1[li], L);
    k_sample_m<<<dim3(L, NHEAD), 64, 0, stream>>>(Qb, Kb, IDX, MM, L, L, U);
    k_topk<<<NHEAD, 256, 0, stream>>>(MM, MW, MTOP, L, U);
    k_gather_q<<<dim3(U, NHEAD), 64, 0, stream>>>(Qb, MTOP, QR, U);
    // scores[h] = Qr[h] @ k[h]^T / 8 : batch over heads
    gemm(stream, QR, Kb, nullptr, nullptr, SC, U, L, HDIM, HDIM, 512, L,
         (long long)U * HDIM, HDIM, (long long)U * L, 1, 0, 0, 0.125f, NHEAD);
    k_softmax<<<NHEAD * U, 256, 0, stream>>>(SC, L, nullptr);
    gemm(stream, SC, Vb, nullptr, nullptr, UPD, U, HDIM, L, L, 512, HDIM,
         (long long)U * L, HDIM, (long long)U * HDIM, 0, 0, 0, 1.f, NHEAD);
    k_colmean<<<D_MODEL, 256, 0, stream>>>(Vb, MEANS, L);
    k_bcast<<<CDIV(L * D_MODEL, 256), 256, 0, stream>>>(MEANS, T1, L * D_MODEL);
    k_scatter<<<dim3(U, NHEAD), 64, 0, stream>>>(T1, MTOP, UPD, U);

    gemm(stream, T1, wo, bo, ENCX, T2, L, 512, 512, 512, 512, 512, 0,0,0, 0,0,0, 1.f, 1);
    k_ln<<<L, 256, 0, stream>>>(T2, ln1g, ln1b, ENCX);
    gemm(stream, ENCX, w1, b1, nullptr, T1, L, 512, 512, 512, 512, 512, 0,0,0, 0,1,0, 1.f, 1);
    gemm(stream, T1, w2, b2, ENCX, T2, L, 512, 512, 512, 512, 512, 0,0,0, 0,0,0, 1.f, 1);
    k_ln<<<L, 256, 0, stream>>>(T2, ln2g, ln2b, ENCX);

    if (li < 2) {  // distill conv + BN + ELU + maxpool(3,2)
      int cb = 61 + 4 * li;
      const float *cbias = IN(cb + 0), *cbeta = IN(cb + 1), *cg = IN(cb + 2), *cw = IN(cb + 3);
      k_aug3<<<CDIV(L * 1536, 256), 256, 0, stream>>>(ENCX, AUG, L, 512);
      gemm(stream, AUG, cw, cbias, nullptr, T1, L, 512, 1536, 1536, 512, 512,
           0, 0, 0, 1, 0, 0, 1.f, 1);
      k_colstats<<<D_MODEL, 256, 0, stream>>>(T1, STATS, L);
      k_bn_elu<<<CDIV(L * D_MODEL, 256), 256, 0, stream>>>(T1, STATS, cg, cbeta, L * D_MODEL);
      int Lo = L / 2;
      k_maxpool<<<CDIV(Lo * D_MODEL, 256), 256, 0, stream>>>(T1, ENCX, L, Lo);
      L = Lo;
    }
  }
  k_ln<<<L, 256, 0, stream>>>(ENCX, IN(121), IN(120), ENCO);
  const int Lenc = L;  // 1024

  // ---- decoder embedding ----
  k_aug3<<<CDIV(L0 * 576, 256), 256, 0, stream>>>(XD, AUG, L0, 192);
  gemm(stream, XMD, IN(6), IN(5), nullptr, T1, L0, 512, 96, 96, 512, 512,
       0, 0, 0, 0, 0, 0, 1.f, 1);
  gemm(stream, AUG, IN(4), nullptr, T1, DECX, L0, 512, 576, 576, 512, 512,
       0, 0, 0, 1, 0, 1, 1.f, 1);

  // ---- decoder stack ----
  const int Ld = L0;
  for (int lj = 0; lj < 2; ++lj) {
    int pb = 7 + 26 * lj;
    const float *b1 = IN(pb + 0), *b2 = IN(pb + 1);
    const float *cbk = IN(pb + 2), *cbo = IN(pb + 3), *cbq = IN(pb + 4), *cbv = IN(pb + 5);
    const float *cwk = IN(pb + 6), *cwo = IN(pb + 7), *cwq = IN(pb + 8), *cwv = IN(pb + 9);
    const float *ln1b = IN(pb + 10), *ln1g = IN(pb + 11);
    const float *ln2b = IN(pb + 12), *ln2g = IN(pb + 13);
    const float *ln3b = IN(pb + 14), *ln3g = IN(pb + 15);
    const float *sbk = IN(pb + 16), *sbo = IN(pb + 17), *sbq = IN(pb + 18), *sbv = IN(pb + 19);
    const float *swk = IN(pb + 20), *swo = IN(pb + 21), *swq = IN(pb + 22), *swv = IN(pb + 23);
    const float *w1 = IN(pb + 24), *w2 = IN(pb + 25);

    // self ProbSparse attention (causal, mix=True)
    gemm(stream, DECX, swq, sbq, nullptr, Qb, Ld, 512, 512, 512, 512, 512, 0,0,0, 0,0,0, 1.f, 1);
    gemm(stream, DECX, swk, sbk, nullptr, Kb, Ld, 512, 512, 512, 512, 512, 0,0,0, 0,0,0, 1.f, 1);
    gemm(stream, DECX, swv, sbv, nullptr, Vb, Ld, 512, 512, 512, 512, 512, 0,0,0, 0,0,0, 1.f, 1);
    int U = probU(Ld);  // 45
    k_randidx<<<CDIV(Ld * U, 256), 256, 0, stream>>>(IDX, Ld * U, key0[3 + lj], key1[3 + lj], Ld);
    k_sample_m<<<dim3(Ld, NHEAD), 64, 0, stream>>>(Qb, Kb, IDX, MM, Ld, Ld, U);
    k_topk<<<NHEAD, 256, 0, stream>>>(MM, MW, MTOP, Ld, U);
    k_gather_q<<<dim3(U, NHEAD), 64, 0, stream>>>(Qb, MTOP, QR, U);
    gemm(stream, QR, Kb, nullptr, nullptr, SC, U, Ld, HDIM, HDIM, 512, Ld,
         (long long)U * HDIM, HDIM, (long long)U * Ld, 1, 0, 0, 0.125f, NHEAD);
    k_softmax<<<NHEAD * U, 256, 0, stream>>>(SC, Ld, MTOP);  // causal: col <= Mtop
    gemm(stream, SC, Vb, nullptr, nullptr, UPD, U, HDIM, Ld, Ld, 512, HDIM,
         (long long)U * Ld, HDIM, (long long)U * HDIM, 0, 0, 0, 1.f, NHEAD);
    k_cumsum<<<2, 256, 0, stream>>>(Vb, T1, Ld);
    k_scatter<<<dim3(U, NHEAD), 64, 0, stream>>>(T1, MTOP, UPD, U);
    k_mix<<<CDIV(Ld * D_MODEL, 256), 256, 0, stream>>>(T1, T2, Ld);
    gemm(stream, T2, swo, sbo, DECX, T1, Ld, 512, 512, 512, 512, 512, 0,0,0, 0,0,0, 1.f, 1);
    k_ln<<<Ld, 256, 0, stream>>>(T1, ln1g, ln1b, DECX);

    // cross full attention (per-head to bound scratch)
    gemm(stream, DECX, cwq, cbq, nullptr, Qb, Ld, 512, 512, 512, 512, 512, 0,0,0, 0,0,0, 1.f, 1);
    gemm(stream, ENCO, cwk, cbk, nullptr, Kb, Lenc, 512, 512, 512, 512, 512, 0,0,0, 0,0,0, 1.f, 1);
    gemm(stream, ENCO, cwv, cbv, nullptr, Vb, Lenc, 512, 512, 512, 512, 512, 0,0,0, 0,0,0, 1.f, 1);
    for (int h = 0; h < NHEAD; ++h) {
      gemm(stream, Qb + h * HDIM, Kb + h * HDIM, nullptr, nullptr, SC,
           Ld, Lenc, HDIM, 512, 512, Lenc, 0, 0, 0, 1, 0, 0, 0.125f, 1);
      k_softmax<<<Ld, 256, 0, stream>>>(SC, Lenc, nullptr);
      gemm(stream, SC, Vb + h * HDIM, nullptr, nullptr, T1 + h * HDIM,
           Ld, HDIM, Lenc, Lenc, 512, 512, 0, 0, 0, 0, 0, 0, 1.f, 1);
    }
    gemm(stream, T1, cwo, cbo, DECX, T2, Ld, 512, 512, 512, 512, 512, 0,0,0, 0,0,0, 1.f, 1);
    k_ln<<<Ld, 256, 0, stream>>>(T2, ln2g, ln2b, DECX);

    // FFN
    gemm(stream, DECX, w1, b1, nullptr, T1, Ld, 512, 512, 512, 512, 512, 0,0,0, 0,1,0, 1.f, 1);
    gemm(stream, T1, w2, b2, DECX, T2, Ld, 512, 512, 512, 512, 512, 0,0,0, 0,0,0, 1.f, 1);
    k_ln<<<Ld, 256, 0, stream>>>(T2, ln3g, ln3b, DECX);
  }

  // ---- final norm + projection + skip head ----
  k_ln<<<Ld, 256, 0, stream>>>(DECX, IN(60), IN(59), T1);
  k_proj<<<Ld, 256, 0, stream>>>(T1, IN(123), IN(122), XD, IN(125), IN(124),
                                 (float*)d_out);
#undef IN
}
